// MeMA_4166118277534
// MI455X (gfx1250) — matrix operand order repaired and checked by reading, not compiled, not run on hardware
//
#include <hip/hip_runtime.h>
#include <hip/hip_bf16.h>

// ---------------------------------------------------------------------------
// MeMA dual-branch memory attention for MI455X (gfx1250, wave32, WMMA).
// All GEMMs via v_wmma_f32_16x16x32_bf16 (bf16 in, f32 accumulate).
// Branch dimension in the grid keeps VGPR use < 256 (no s_set_vgpr_msb churn).
// ---------------------------------------------------------------------------

typedef __attribute__((ext_vector_type(16))) __bf16 v16bf;
typedef __attribute__((ext_vector_type(8)))  __bf16 v8bf;
typedef __attribute__((ext_vector_type(8)))  float  v8f;

#define WMMA_BF16(A, B, C) \
  __builtin_amdgcn_wmma_f32_16x16x32_bf16(false, (A), false, (B), (short)0, (C), false, false)

// Problem constants
#define BATCH 64
#define CCH   128      // channels per head
#define HEADS 8
#define NTOK  256      // n = h*w = M
#define KDIM  1024     // heads * C

// Workspace layout (bytes)
#define FFLAT_OFF 0u                      // [B][N][C] bf16        :  4,194,304
#define WT_OFF    4194304u                // [4][H][co128][ci128]  :  1,048,576
#define WOT_OFF   5242880u                // [co128][ki1024] bf16  :    262,144
#define OBUF_OFF  5505024u                // 2 x [B][N][1024] bf16 : 67,108,864
#define OBR_ELEMS ((size_t)BATCH * NTOK * KDIM)

// ---- A-fragment loader: 16-bit A 16x32, contiguous k order:
// lane holds row = lane%16; halfwords 0..15 = K = (lane/16)*16 .. +15.
__device__ inline v16bf make_a(const __bf16* __restrict__ row, int kbase, int lh) {
  v8bf lo = *(const v8bf*)(row + kbase + lh * 16);
  v8bf hi = *(const v8bf*)(row + kbase + 8 + lh * 16);
  return __builtin_shufflevector(lo, hi, 0,1,2,3,4,5,6,7,8,9,10,11,12,13,14,15);
}

__device__ inline v16bf make_a_f32(const float* __restrict__ row, int kbase, int lh) {
  v16bf r;
#pragma unroll
  for (int i = 0; i < 8; ++i) {
    r[i]     = (__bf16)row[kbase + lh * 16 + i];
    r[i + 8] = (__bf16)row[kbase + 8 + lh * 16 + i];
  }
  return r;
}

// Sum two bf16 A-fragments in f32 (branch combine), back to bf16.
__device__ inline v16bf add_frag(v16bf a, v16bf b) {
  v16bf r;
#pragma unroll
  for (int i = 0; i < 16; ++i) r[i] = (__bf16)((float)a[i] + (float)b[i]);
  return r;
}

__device__ inline float redmax16(float v) {
#pragma unroll
  for (int m = 1; m < 16; m <<= 1) v = fmaxf(v, __shfl_xor(v, m, 16));
  return v;
}
__device__ inline float redsum16(float v) {
#pragma unroll
  for (int m = 1; m < 16; m <<= 1) v += __shfl_xor(v, m, 16);
  return v;
}

// ---------------------------------------------------------------------------
// Prep kernels: f32 -> bf16 with layout changes so WMMA B-frags are contiguous.
// ---------------------------------------------------------------------------
__global__ void prep_fflat(const float* __restrict__ f, __bf16* __restrict__ ff) {
  int t = blockIdx.x * 256 + threadIdx.x;     // B*C*N = 2,097,152
  int b = t >> 15;                            // /(128*256)
  int r = t & 32767;
  int c = r >> 8;
  int n = r & 255;
  ff[(size_t)(b * NTOK + n) * CCH + c] = (__bf16)f[t];
}

// WT[mat][h][co][ci] = Wsrc[mat][ci][h*128+co]; mat order: k2g, v2g, k2t, v2t
__global__ void prep_w(const float* __restrict__ a, const float* __restrict__ b,
                       const float* __restrict__ c, const float* __restrict__ d,
                       __bf16* __restrict__ WT) {
  int t = blockIdx.x * 256 + threadIdx.x;     // 4*128*1024 = 524,288
  int mi = t >> 17;
  int r  = t & 131071;
  int ci = r >> 10;
  int o  = r & 1023;
  const float* src = (mi == 0) ? a : (mi == 1) ? b : (mi == 2) ? c : d;
  int h = o >> 7, co = o & 127;
  WT[((size_t)((mi * HEADS + h) * CCH + co)) * CCH + ci] = (__bf16)src[r];
}

__global__ void prep_wo(const float* __restrict__ wo, __bf16* __restrict__ WoT) {
  int t = blockIdx.x * 256 + threadIdx.x;     // 1024*128 = 131,072
  int ki = t >> 7, co = t & 127;
  WoT[(size_t)co * KDIM + ki] = (__bf16)wo[t];
}

// ---------------------------------------------------------------------------
// Fused projection + single-branch attention.
// Grid: B*H*2(mhalf)*2(br). Block: 256 threads = 8 wave32, wave = 16 q-rows.
// Dynamic LDS: ldsK 64KB (K matrix, then probs), ldsVt 8KB (transposed V stripe).
// ---------------------------------------------------------------------------
__global__ __launch_bounds__(256) void mema_attn(
    const __bf16* __restrict__ fflat, const __bf16* __restrict__ WT,
    const float* __restrict__ mem, const int* __restrict__ task_ids,
    __bf16* __restrict__ Obuf) {
  extern __shared__ __align__(128) __bf16 smem[];
  __bf16* ldsK  = smem;                 // [256][128] / probs [8][16][256]
  __bf16* ldsVt = smem + NTOK * CCH;    // [128][32] (channel-major)

  const int tid = threadIdx.x;
  const int w = tid >> 5, lane = tid & 31;
  const int l16 = lane & 15, lh = lane >> 4;
  const int wg = blockIdx.x;
  const int br = wg & 1, mhalf = (wg >> 1) & 1, h = (wg >> 2) & 7, b = wg >> 5;
  const int mrow0 = (mhalf * 8 + w) * 16;
  const __bf16* fb = fflat + (size_t)b * NTOK * CCH;

  const v8f vz = {0.f, 0.f, 0.f, 0.f, 0.f, 0.f, 0.f, 0.f};

  const __bf16* Wk = WT + (size_t)((br * 2 + 0) * HEADS + h) * CCH * CCH;
  const __bf16* Wv = WT + (size_t)((br * 2 + 1) * HEADS + h) * CCH * CCH;
  const int qidx = (br == 0) ? 4 : task_ids[b];             // global uses mem[-1]
  const float* q = mem + (size_t)qidx * (CCH * NTOK * HEADS) + (size_t)h * (NTOK * CCH);
  __bf16* Ob = Obuf + (size_t)br * OBR_ELEMS;

  // ---- Stage 1: K = fflat @ Wk_h  ->  ldsK[n][c] (bf16)
#pragma unroll
  for (int ntl = 0; ntl < 2; ++ntl) {
    const int nt = 2 * w + ntl;
    const __bf16* arow = fb + (nt * 16 + l16) * CCH;
    v16bf afr[4];
#pragma unroll
    for (int kc = 0; kc < 4; ++kc) afr[kc] = make_a(arow, kc * 32, lh);
#pragma unroll
    for (int ct = 0; ct < 8; ++ct) {
      v8f acc = vz;
#pragma unroll
      for (int kc = 0; kc < 4; ++kc) {
        v16bf bfr = *(const v16bf*)(Wk + (ct * 16 + l16) * CCH + kc * 32 + lh * 16);
        acc = WMMA_BF16(afr[kc], bfr, acc);
      }
#pragma unroll
      for (int e = 0; e < 8; ++e)
        ldsK[(nt * 16 + e + 8 * lh) * CCH + ct * 16 + l16] = (__bf16)acc[e];
    }
  }
  __syncthreads();

  // ---- Stage 2: scores = q @ K^T (16x256 per wave), softmax in registers
  {
    v8f sacc[16];
#pragma unroll
    for (int nt = 0; nt < 16; ++nt) sacc[nt] = vz;
#pragma unroll
    for (int kc = 0; kc < 4; ++kc) {
      v16bf aq = make_a_f32(q + (mrow0 + l16) * CCH, kc * 32, lh);
#pragma unroll
      for (int nt = 0; nt < 16; ++nt) {
        v16bf bk = *(const v16bf*)(ldsK + (nt * 16 + l16) * CCH + kc * 32 + lh * 16);
        sacc[nt] = WMMA_BF16(aq, bk, sacc[nt]);
      }
    }
    float mx[8], sm[8];
#pragma unroll
    for (int e = 0; e < 8; ++e) mx[e] = -3.4e38f;
#pragma unroll
    for (int nt = 0; nt < 16; ++nt)
#pragma unroll
      for (int e = 0; e < 8; ++e) mx[e] = fmaxf(mx[e], sacc[nt][e]);
#pragma unroll
    for (int e = 0; e < 8; ++e) { mx[e] = redmax16(mx[e]); sm[e] = 0.f; }
#pragma unroll
    for (int nt = 0; nt < 16; ++nt)
#pragma unroll
      for (int e = 0; e < 8; ++e) {
        float p = __expf(sacc[nt][e] - mx[e]);
        sacc[nt][e] = p;
        sm[e] += p;
      }
#pragma unroll
    for (int e = 0; e < 8; ++e) sm[e] = 1.f / redsum16(sm[e]);
    __syncthreads();  // all waves done reading ldsK

    // ---- Stage 3: probabilities -> ldsK (wave-private [16][256] bf16 slab)
#pragma unroll
    for (int nt = 0; nt < 16; ++nt)
#pragma unroll
      for (int e = 0; e < 8; ++e)
        ldsK[w * 4096 + (e + 8 * lh) * 256 + nt * 16 + l16] =
            (__bf16)(sacc[nt][e] * sm[e]);
    __syncthreads();
  }

  // ---- Stage 4: stream 32-row V stripes; o += P_chunk @ V_chunk
  v8f oacc[8];
#pragma unroll
  for (int ct = 0; ct < 8; ++ct) oacc[ct] = vz;

  for (int s8 = 0; s8 < 8; ++s8) {
#pragma unroll
    for (int ntl = 0; ntl < 2; ++ntl) {
      const int n0 = s8 * 32 + ntl * 16;
      const __bf16* arow = fb + (n0 + l16) * CCH;
      v8f acc = vz;
#pragma unroll
      for (int kc = 0; kc < 4; ++kc) {
        v16bf a   = make_a(arow, kc * 32, lh);
        v16bf bfr = *(const v16bf*)(Wv + (w * 16 + l16) * CCH + kc * 32 + lh * 16);
        acc = WMMA_BF16(a, bfr, acc);
      }
      // store TRANSPOSED: ldsVt[channel][row-in-stripe]
#pragma unroll
      for (int e = 0; e < 8; ++e)
        ldsVt[(w * 16 + l16) * 32 + ntl * 16 + e + 8 * lh] = (__bf16)acc[e];
    }
    __syncthreads();
    v16bf ap = make_a(ldsK + w * 4096 + l16 * 256, s8 * 32, lh);
#pragma unroll
    for (int ct = 0; ct < 8; ++ct) {
      v16bf bv = *(const v16bf*)(ldsVt + (ct * 16 + l16) * 32 + lh * 16);
      oacc[ct] = WMMA_BF16(ap, bv, oacc[ct]);
    }
    __syncthreads();
  }

  // ---- Write this branch's o to O_br[b][n][h*128+c] bf16
#pragma unroll
  for (int ct = 0; ct < 8; ++ct)
#pragma unroll
    for (int e = 0; e < 8; ++e)
      Ob[(size_t)(b * NTOK + mrow0 + e + 8 * lh) * KDIM + h * CCH + ct * 16 + l16] =
          (__bf16)oacc[ct][e];
}

// ---------------------------------------------------------------------------
// Output projection (sum of both branch buffers) + bias + LayerNorm +
// transpose to [b][c][n] f32.
// Grid: 128 blocks * 8 waves; each wave = one (b, 16-token tile).
// ---------------------------------------------------------------------------
__global__ __launch_bounds__(256) void mema_out(
    const __bf16* __restrict__ Obuf, const __bf16* __restrict__ WoT,
    const float* __restrict__ bo, float* __restrict__ out) {
  const int tid = threadIdx.x;
  const int w = tid >> 5, lane = tid & 31, l16 = lane & 15, lh = lane >> 4;
  const int wid = blockIdx.x * 8 + w;
  const int b = wid >> 4, nt = wid & 15;
  const __bf16* arow0 = Obuf + (size_t)(b * NTOK + nt * 16 + l16) * KDIM;
  const __bf16* arow1 = arow0 + OBR_ELEMS;

  const v8f vz = {0.f, 0.f, 0.f, 0.f, 0.f, 0.f, 0.f, 0.f};
  v8f acc[8];
#pragma unroll
  for (int ct = 0; ct < 8; ++ct) acc[ct] = vz;

  for (int kc = 0; kc < 32; ++kc) {
    v16bf a = add_frag(make_a(arow0, kc * 32, lh), make_a(arow1, kc * 32, lh));
#pragma unroll
    for (int ct = 0; ct < 8; ++ct) {
      v16bf bfr = *(const v16bf*)(WoT + (size_t)(ct * 16 + l16) * KDIM + kc * 32 + lh * 16);
      acc[ct] = WMMA_BF16(a, bfr, acc[ct]);
    }
  }
  // bias
#pragma unroll
  for (int ct = 0; ct < 8; ++ct) {
    float bv = bo[ct * 16 + l16];
#pragma unroll
    for (int e = 0; e < 8; ++e) acc[ct][e] += bv;
  }
  // LayerNorm over channels (biased variance, eps = 1e-5)
  float s1[8], s2[8];
#pragma unroll
  for (int e = 0; e < 8; ++e) { s1[e] = 0.f; s2[e] = 0.f; }
#pragma unroll
  for (int ct = 0; ct < 8; ++ct)
#pragma unroll
    for (int e = 0; e < 8; ++e) {
      float x = acc[ct][e];
      s1[e] += x;
      s2[e] += x * x;
    }
#pragma unroll
  for (int e = 0; e < 8; ++e) {
    float mean = redsum16(s1[e]) * (1.f / 128.f);
    float var  = redsum16(s2[e]) * (1.f / 128.f) - mean * mean;
    s1[e] = mean;
    s2[e] = rsqrtf(var + 1e-5f);
  }
  // store transposed: out[b][c][n_tok]
#pragma unroll
  for (int ct = 0; ct < 8; ++ct)
#pragma unroll
    for (int e = 0; e < 8; ++e) {
      int ntok = nt * 16 + e + 8 * lh;
      int c    = ct * 16 + l16;
      out[((size_t)(b * CCH + c)) * NTOK + ntok] = (acc[ct][e] - s1[e]) * s2[e];
    }
}

// ---------------------------------------------------------------------------
extern "C" void kernel_launch(void* const* d_in, const int* in_sizes, int n_in,
                              void* d_out, int out_size, void* d_ws, size_t ws_size,
                              hipStream_t stream) {
  const float* feature  = (const float*)d_in[0];
  const int*   task_ids = (const int*)d_in[1];
  const float* mem      = (const float*)d_in[2];
  const float* Wk2g     = (const float*)d_in[3];
  const float* Wv2g     = (const float*)d_in[4];
  const float* Wk2t     = (const float*)d_in[5];
  const float* Wv2t     = (const float*)d_in[6];
  const float* Wo       = (const float*)d_in[7];
  const float* bo       = (const float*)d_in[8];

  char* ws = (char*)d_ws;
  __bf16* fflat = (__bf16*)(ws + FFLAT_OFF);
  __bf16* WT    = (__bf16*)(ws + WT_OFF);
  __bf16* WoT   = (__bf16*)(ws + WOT_OFF);
  __bf16* Obuf  = (__bf16*)(ws + OBUF_OFF);

  prep_fflat<<<8192, 256, 0, stream>>>(feature, fflat);
  prep_w<<<2048, 256, 0, stream>>>(Wk2g, Wv2g, Wk2t, Wv2t, WT);
  prep_wo<<<512, 256, 0, stream>>>(Wo, WoT);

  // 8 waves/block, 73728 B dynamic LDS (K/probs 64KB + V-stripe 8KB)
  mema_attn<<<BATCH * HEADS * 2 * 2, 256, 73728, stream>>>(fflat, WT, mem, task_ids, Obuf);

  mema_out<<<128, 256, 0, stream>>>(Obuf, WoT, bo, (float*)d_out);
}
